// sLSTMBlock_70428873719992
// MI455X (gfx1250) — matrix operand order repaired and checked by reading, not compiled, not run on hardware
//
#include <hip/hip_runtime.h>
#include <hip/hip_bf16.h>

// ---------------- problem constants ----------------
constexpr int B  = 8;
constexpr int T  = 2048;
constexpr int D  = 1024;
constexpr int NH = 8;
constexpr int DH = 128;
constexpr int H  = NH * DH;       // 1024
constexpr int P  = 1365;          // int(4/3 * H)
constexpr int P2 = 2 * P;         // 2730
constexpr int PPAD  = 1376;       // pad P to 43*32 (multiple of 32 for WMMA K)
constexpr int P2PAD = 2816;       // pad 2P to 176*16
constexpr float EPS = 1e-6f;
constexpr float CAP = 15.0f;

constexpr int NWG_SCAN = 16;      // one workgroup-cluster worth of WGs
constexpr int NTHR     = 256;     // 8 wave32 per WG

// ---------------- WMMA / vector types ----------------
typedef __bf16 bf16_t;
typedef __attribute__((ext_vector_type(16))) __bf16 v16bf;
typedef __attribute__((ext_vector_type(8)))  float  v8f;
typedef unsigned int u32x4 __attribute__((ext_vector_type(4)));
typedef int          i32x4 __attribute__((ext_vector_type(4)));
typedef int          i32x8 __attribute__((ext_vector_type(8)));

#if __has_builtin(__builtin_amdgcn_tensor_load_to_lds)
#define HAVE_TDM 1
#else
#define HAVE_TDM 0
#endif

__device__ __forceinline__ v8f wmma_bf16(v16bf a, v16bf b, v8f c) {
  // D = A(16x32 bf16) * B(32x16 bf16) + C(16x16 f32)
  return __builtin_amdgcn_wmma_f32_16x16x32_bf16(
      /*neg_a=*/false, a, /*neg_b=*/false, b,
      /*c_mod=*/(short)0, c, /*reuse_a=*/false, /*reuse_b=*/false);
}

__device__ __forceinline__ bf16_t f2bf(float f) {
  union { float fv; unsigned uv; } c; c.fv = f;
  unsigned r = (c.uv + 0x7FFFu + ((c.uv >> 16) & 1u)) >> 16;  // RNE
  union { unsigned short s; bf16_t b; } o; o.s = (unsigned short)r;
  return o.b;
}

#if HAVE_TDM
// Build a 2D bf16 tile descriptor (D#) and issue TENSOR_LOAD_TO_LDS.
// Tile: tile_x contiguous elements per row, tile_y rows, row pitch
// row_stride elements. LDS dst receives rows back-to-back.
// ISA D# layout: CDNA5 ISA.
// This toolchain declares the 6-arg builtin:
//   (u32x4 g0, i32x8 g1, i32x4 g2, i32x4 g3, i32x8 extra, i32 cpol)
__device__ __forceinline__ void tdm_load_2d(const bf16_t* gsrc, bf16_t* ldst,
                                            unsigned tile_x, unsigned tile_y,
                                            unsigned long long row_stride) {
  unsigned long long ga = (unsigned long long)gsrc;
  // flat LDS pointer: low 32 bits are the LDS byte offset (ISA 10.2 aperture)
  unsigned la = (unsigned)(unsigned long long)(const void*)ldst;
  u32x4 g0;
  g0[0] = 1u;                                          // count=1, user mode
  g0[1] = la;                                          // lds_addr
  g0[2] = (unsigned)(ga & 0xFFFFFFFFull);              // global_addr[31:0]
  g0[3] = (unsigned)((ga >> 32) & 0x1FFFFFFu) | (2u << 30);  // addr[56:32], type=2
  const unsigned tdx = 1u << 20, tdy = 1u << 20;       // generous tensor dims (in-bounds tile)
  i32x8 g1;
  g1[0] = 0x00010000;                                  // wg_mask=0, data_size=1 (2B)
  g1[1] = (int)((tdx & 0xFFFFu) << 16);                // tensor_dim0[15:0] @63:48
  g1[2] = (int)((tdx >> 16) | ((tdy & 0xFFFFu) << 16));// dim0 hi16 | dim1 lo16
  g1[3] = (int)((tdy >> 16) | (tile_x << 16));         // dim1 hi16 | tile_dim0
  g1[4] = (int)(tile_y);                               // tile_dim1 (tile_dim2=0)
  g1[5] = (int)(unsigned)(row_stride & 0xFFFFFFFFull); // dim0_stride[31:0]
  g1[6] = (int)(unsigned)((row_stride >> 32) & 0xFFFFull); // stride hi16 | dim1_stride lo=0
  g1[7] = 0;
  i32x4 z4 = { 0, 0, 0, 0 };
  i32x8 z8 = { 0, 0, 0, 0, 0, 0, 0, 0 };
  __builtin_amdgcn_tensor_load_to_lds(g0, g1, z4, z4, z8, 0);
}
#endif

// ---------------- workspace layout (bytes) ----------------
constexpr size_t AL(size_t x) { return (x + 255) & ~(size_t)255; }
constexpr size_t SZ_XN   = (size_t)B * T * D * 2;          // bf16 x_n
constexpr size_t SZ_GATE = (size_t)B * T * H * 4;          // one f32 gate plane
constexpr size_t SZ_WB   = (size_t)4 * D * H * 2;          // bf16 Wz|Wi|Wo|Wf
constexpr size_t SZ_RB   = (size_t)4 * NH * DH * DH * 2;   // bf16 Rz|Ri|Ro|Rf
constexpr size_t SZ_UPB  = (size_t)H * P2PAD * 2;          // bf16 up_w (N padded)
constexpr size_t SZ_DNB  = (size_t)PPAD * D * 2;           // bf16 down_w (K padded)
constexpr size_t SZ_ST   = (size_t)B * H * 4;              // one f32 state plane
constexpr size_t SZ_HB   = (size_t)NH * 16 * DH * 2;       // bf16 h, M padded to 16
constexpr size_t SZ_RPRE = (size_t)4 * B * H * 4;          // f32 recurrent preacts
constexpr size_t SZ_OUTB = (size_t)16 * H * 2;             // bf16 mhln(h), M padded
constexpr size_t SZ_U    = (size_t)16 * P2PAD * 4;         // f32 up-proj result
constexpr size_t SZ_GACT = (size_t)16 * PPAD * 2;          // bf16 u1*gelu(u2)

constexpr size_t OFF_XN   = 0;
constexpr size_t OFF_G    = AL(OFF_XN + SZ_XN);            // 4 planes
constexpr size_t OFF_WB   = AL(OFF_G + 4 * SZ_GATE);
constexpr size_t OFF_RB   = AL(OFF_WB + SZ_WB);
constexpr size_t OFF_UPB  = AL(OFF_RB + SZ_RB);
constexpr size_t OFF_DNB  = AL(OFF_UPB + SZ_UPB);
constexpr size_t OFF_C    = AL(OFF_DNB + SZ_DNB);
constexpr size_t OFF_N    = AL(OFF_C + SZ_ST);
constexpr size_t OFF_HF   = AL(OFF_N + SZ_ST);
constexpr size_t OFF_M    = AL(OFF_HF + SZ_ST);
constexpr size_t OFF_HB   = AL(OFF_M + SZ_ST);
constexpr size_t OFF_RPRE = AL(OFF_HB + SZ_HB);
constexpr size_t OFF_OUTB = AL(OFF_RPRE + SZ_RPRE);
constexpr size_t OFF_U    = AL(OFF_OUTB + SZ_OUTB);
constexpr size_t OFF_GACT = AL(OFF_U + SZ_U);
constexpr size_t OFF_SYNC = AL(OFF_GACT + SZ_GACT);

// ---------------- init: states + sync + padded buffers ----------------
__global__ void k_init(float* c, float* n, float* hf, float* m,
                       bf16_t* hb, bf16_t* outb, bf16_t* gact, unsigned* sync) {
  int i = blockIdx.x * blockDim.x + threadIdx.x;
  int stride = gridDim.x * blockDim.x;
  const bf16_t z = f2bf(0.0f);
  for (int e = i; e < B * H; e += stride) { c[e] = 0.f; n[e] = 1.f; hf[e] = 0.f; m[e] = 0.f; }
  for (int e = i; e < NH * 16 * DH; e += stride) hb[e] = z;
  for (int e = i; e < 16 * H; e += stride) outb[e] = z;
  for (int e = i; e < 16 * PPAD; e += stride) gact[e] = z;
  if (i < 64) sync[i] = 0u;
}

// ---------------- weight conversion to bf16 (with padding) ----------------
__global__ void k_prep(const float* Wz, const float* Wi, const float* Wo, const float* Wf,
                       const float* Rz, const float* Ri, const float* Ro, const float* Rf,
                       const float* up_w, const float* down_w,
                       bf16_t* Wb, bf16_t* Rb, bf16_t* upb, bf16_t* dnb) {
  long i = (long)blockIdx.x * blockDim.x + threadIdx.x;
  long stride = (long)gridDim.x * blockDim.x;
  const long NW = 4L * D * H, NR = 4L * NH * DH * DH;
  const long NU = (long)H * P2PAD, ND = (long)PPAD * D;
  for (long e = i; e < NW; e += stride) {
    int g = (int)(e / ((long)D * H)); long r = e % ((long)D * H);
    const float* s = (g == 0) ? Wz : (g == 1) ? Wi : (g == 2) ? Wo : Wf;
    Wb[e] = f2bf(s[r]);
  }
  for (long e = i; e < NR; e += stride) {
    int g = (int)(e / ((long)NH * DH * DH)); long r = e % ((long)NH * DH * DH);
    const float* s = (g == 0) ? Rz : (g == 1) ? Ri : (g == 2) ? Ro : Rf;
    Rb[e] = f2bf(s[r]);
  }
  for (long e = i; e < NU; e += stride) {
    long d = e / P2PAD, col = e % P2PAD;
    upb[e] = f2bf(col < P2 ? up_w[d * P2 + col] : 0.0f);
  }
  for (long e = i; e < ND; e += stride) {
    long krow = e / D, col = e % D;
    dnb[e] = f2bf(krow < P ? down_w[krow * D + col] : 0.0f);
  }
}

// ---------------- LayerNorm(x) -> bf16 x_n  (one block per row) ----------------
__global__ void __launch_bounds__(NTHR) k_ln(const float* __restrict__ x,
                                             const float* __restrict__ w,
                                             const float* __restrict__ b,
                                             bf16_t* __restrict__ xn) {
  const long row = blockIdx.x;                         // 0 .. B*T-1
  const float* xr = x + row * D;
  float4 v = ((const float4*)xr)[threadIdx.x];         // 256 thr * 4 = 1024
  float s  = v.x + v.y + v.z + v.w;
  float ss = v.x * v.x + v.y * v.y + v.z * v.z + v.w * v.w;
  #pragma unroll
  for (int off = 16; off > 0; off >>= 1) {
    s  += __shfl_xor(s, off, 32);
    ss += __shfl_xor(ss, off, 32);
  }
  __shared__ float sw[8], sw2[8];
  int wave = threadIdx.x >> 5, lane = threadIdx.x & 31;
  if (lane == 0) { sw[wave] = s; sw2[wave] = ss; }
  __syncthreads();
  if (threadIdx.x == 0) {
    float a = 0.f, a2 = 0.f;
    for (int k = 0; k < 8; ++k) { a += sw[k]; a2 += sw2[k]; }
    sw[0] = a; sw2[0] = a2;
  }
  __syncthreads();
  float mu = sw[0] * (1.0f / D);
  float var = sw2[0] * (1.0f / D) - mu * mu;
  float rstd = rsqrtf(var + EPS);
  int c0 = threadIdx.x * 4;
  float vals[4] = { v.x, v.y, v.z, v.w };
  #pragma unroll
  for (int k = 0; k < 4; ++k)
    xn[row * D + c0 + k] = f2bf((vals[k] - mu) * rstd * w[c0 + k] + b[c0 + k]);
}

// ---------------- gate GEMMs: gate[g] = x_n @ W[g] + b[g] ----------------
// grid = (M/128, H/128, 4 gates), block = 256 (8 waves; 4x2 wave tiling, 32x64 / wave)
// TDM path: double-buffered LDS tiles DMA'd by the Tensor Data Mover while
// all 8 waves run WMMAs; wave 0 issues descriptors + s_wait_tensorcnt.
__global__ void __launch_bounds__(NTHR) k_gemm_gates(const bf16_t* __restrict__ xn,
                                                     const bf16_t* __restrict__ Wb,
                                                     const float* bz, const float* bi,
                                                     const float* bo, const float* bfg,
                                                     float* __restrict__ gates) {
  const int g = blockIdx.z;
  const float* bias = (g == 0) ? bz : (g == 1) ? bi : (g == 2) ? bo : bfg;
  float* out = gates + (long)g * B * T * H;
  const bf16_t* W = Wb + (long)g * D * H;

  const int tid = threadIdx.x;
  const int m0 = blockIdx.x * 128;
  const int n0 = blockIdx.y * 128;
  const int wave = tid >> 5, lane = tid & 31;
  const int wr = wave & 3, wc = wave >> 2;    // wave tile: rows wr*32, cols wc*64

  const v8f vzero = { 0.f, 0.f, 0.f, 0.f, 0.f, 0.f, 0.f, 0.f };
  v8f acc[2][4];
  #pragma unroll
  for (int mt = 0; mt < 2; ++mt)
    #pragma unroll
    for (int nt = 0; nt < 4; ++nt) acc[mt][nt] = vzero;

#if HAVE_TDM
  __shared__ bf16_t ldsA[2][128 * 32];
  __shared__ bf16_t ldsB[2][32 * 128];
  if (wave == 0) {   // prologue: DMA first K-tile pair
    tdm_load_2d(&xn[(long)m0 * D], &ldsA[0][0], 32u, 128u, (unsigned long long)D);
    tdm_load_2d(&W[n0], &ldsB[0][0], 128u, 32u, (unsigned long long)H);
  }
  int cur = 0;
  for (int k0 = 0; k0 < D; k0 += 32) {
    const bool has_next = (k0 + 32 < D);
    if (wave == 0) {
      if (has_next) {   // DMA next pair into the other buffer
        tdm_load_2d(&xn[(long)m0 * D + k0 + 32], &ldsA[cur ^ 1][0], 32u, 128u,
                    (unsigned long long)D);
        tdm_load_2d(&W[(long)(k0 + 32) * H + n0], &ldsB[cur ^ 1][0], 128u, 32u,
                    (unsigned long long)H);
        __builtin_amdgcn_s_wait_tensorcnt(2);  // current pair done, next in flight
      } else {
        __builtin_amdgcn_s_wait_tensorcnt(0);
      }
    }
    __syncthreads();
    v16bf afr[2], bfr[4];
    #pragma unroll
    for (int mt = 0; mt < 2; ++mt)
      afr[mt] = *(const v16bf*)&ldsA[cur][(wr * 32 + mt * 16 + (lane & 15)) * 32 + (lane >> 4) * 16];
    #pragma unroll
    for (int nt = 0; nt < 4; ++nt)
      for (int s = 0; s < 16; ++s) bfr[nt][s] = ldsB[cur][((lane >> 4) * 16 + s) * 128 + wc * 64 + nt * 16 + (lane & 15)];
    #pragma unroll
    for (int mt = 0; mt < 2; ++mt)
      #pragma unroll
      for (int nt = 0; nt < 4; ++nt)
        acc[mt][nt] = wmma_bf16(afr[mt], bfr[nt], acc[mt][nt]);
    __syncthreads();   // all reads of buffer `cur` done before it is re-DMA'd
    cur ^= 1;
  }
#else
  __shared__ bf16_t ldsA[128 * 32];
  __shared__ bf16_t ldsB[32 * 128];
  const int ra = tid >> 1, ca = (tid & 1) * 16;     // A-tile stage: 128x32
  const int rb = tid >> 3, cb = (tid & 7) * 16;     // B-tile stage: 32x128
  for (int k0 = 0; k0 < D; k0 += 32) {
    *(v16bf*)&ldsA[ra * 32 + ca]  = *(const v16bf*)&xn[(long)(m0 + ra) * D + k0 + ca];
    *(v16bf*)&ldsB[rb * 128 + cb] = *(const v16bf*)&W[(long)(k0 + rb) * H + n0 + cb];
    if (k0 + 32 < D) {
      __builtin_prefetch(&xn[(long)(m0 + ra) * D + k0 + 32 + ca], 0, 1);
      __builtin_prefetch(&W[(long)(k0 + 32 + rb) * H + n0 + cb], 0, 1);
    }
    __syncthreads();
    v16bf afr[2], bfr[4];
    #pragma unroll
    for (int mt = 0; mt < 2; ++mt)
      afr[mt] = *(const v16bf*)&ldsA[(wr * 32 + mt * 16 + (lane & 15)) * 32 + (lane >> 4) * 16];
    #pragma unroll
    for (int nt = 0; nt < 4; ++nt)
      for (int s = 0; s < 16; ++s) bfr[nt][s] = ldsB[((lane >> 4) * 16 + s) * 128 + wc * 64 + nt * 16 + (lane & 15)];
    #pragma unroll
    for (int mt = 0; mt < 2; ++mt)
      #pragma unroll
      for (int nt = 0; nt < 4; ++nt)
        acc[mt][nt] = wmma_bf16(afr[mt], bfr[nt], acc[mt][nt]);
    __syncthreads();
  }
#endif

  // C/D layout: VGPR v -> row v (lanes 0-15) / row v+8 (lanes 16-31); col = lane&15
  const int rofs = (lane >= 16) ? 8 : 0;
  #pragma unroll
  for (int mt = 0; mt < 2; ++mt)
    #pragma unroll
    for (int nt = 0; nt < 4; ++nt) {
      int col = n0 + wc * 64 + nt * 16 + (lane & 15);
      float bv = bias[col];
      #pragma unroll
      for (int v = 0; v < 8; ++v) {
        int row = m0 + wr * 32 + mt * 16 + v + rofs;
        out[(long)row * H + col] = acc[mt][nt][v] + bv;
      }
    }
}

// ---------------- persistent scan kernel (16 WGs, cluster-style sync) ----------------
__device__ __forceinline__ void grid_sync(unsigned* cnt, unsigned* gen, unsigned* s_gen) {
  __syncthreads();
  if (threadIdx.x == 0) {
    unsigned g = *s_gen + 1u;
    __threadfence();
    if (atomicAdd(cnt, 1u) == (unsigned)(NWG_SCAN - 1)) {
      atomicExch(cnt, 0u);
      atomicAdd(gen, 1u);
    } else {
      while (atomicAdd(gen, 0u) < g) __builtin_amdgcn_s_sleep(2);
    }
    *s_gen = g;
  }
  __builtin_amdgcn_s_cluster_barrier();  // fast path when launched as a cluster (NOP otherwise)
  __threadfence();
  __syncthreads();
}

__global__ void __launch_bounds__(NTHR, 1) k_scan(
    const float* __restrict__ gz, const float* __restrict__ gi,
    const float* __restrict__ go, const float* __restrict__ gf,
    const bf16_t* __restrict__ Rb,
    const float* rbz, const float* rbi, const float* rbo, const float* rbf,
    const bf16_t* __restrict__ upb, const bf16_t* __restrict__ dnb,
    const float* __restrict__ hln_w, const float* __restrict__ hln_b,
    const float* __restrict__ up_b, const float* __restrict__ down_b,
    const float* __restrict__ x, float* __restrict__ out,
    float* cst, float* nst, float* hst, float* mst,
    bf16_t* hb, float* rpre, bf16_t* outb, float* u, bf16_t* gact,
    unsigned* sync) {
  __shared__ unsigned s_gen;
  __shared__ bf16_t sAct[16 * PPAD];   // A-panel stage for phases D (16xH) and F (16xPPAD)
  if (threadIdx.x == 0) s_gen = 0u;
  __syncthreads();

  (void)__builtin_amdgcn_cluster_id_x();
  __builtin_amdgcn_s_wait_tensorcnt(0);

  const int tid = threadIdx.x, wg = blockIdx.x;
  const int wave = tid >> 5, lane = tid & 31;
  const int gwave = wg * 8 + wave;            // 0..127
  const int gtid = wg * NTHR + tid;           // 0..4095
  unsigned* cnt = sync;
  unsigned* gen = sync + 1;
  const v8f vzero = { 0.f, 0.f, 0.f, 0.f, 0.f, 0.f, 0.f, 0.f };
  const int rl = lane & 15, kh = (lane >> 4) * 16;

  for (int t = 0; t < T; ++t) {
    // -------- Phase A: recurrent block-diag matmuls: rpre[g] = h @ R[g] + rb[g]
    #pragma unroll
    for (int r = 0; r < 2; ++r) {
      int task = gwave + 128 * r;             // 0..255 = (gate,head) x 8 N-tiles
      int pair = task >> 3, ntile = task & 7;
      int g = pair >> 3, head = pair & 7;
      const bf16_t* Rm = Rb + (long)(g * NH + head) * DH * DH;
      v8f acc = vzero;
      #pragma unroll
      for (int k0 = 0; k0 < DH; k0 += 32) {
        v16bf a = *(const v16bf*)&hb[head * (16 * DH) + rl * DH + k0 + kh];
        v16bf bb;
        for (int s = 0; s < 16; ++s) bb[s] = Rm[(long)(k0 + kh + s) * DH + ntile * 16 + rl];
        acc = wmma_bf16(a, bb, acc);
      }
      if (lane < 16) {
        const float* rb = (g == 0) ? rbz : (g == 1) ? rbi : (g == 2) ? rbo : rbf;
        int hidx = head * DH + ntile * 16 + lane;
        float rbv = rb[hidx];
        #pragma unroll
        for (int v = 0; v < B; ++v)
          rpre[(long)g * B * H + (long)v * H + hidx] = acc[v] + rbv;
      }
    }
    grid_sync(cnt, gen, &s_gen);

    // -------- Phase B: gate nonlinearity + state update
    for (int e = gtid; e < B * H; e += NWG_SCAN * NTHR) {
      int b = e >> 10, hidx = e & (H - 1);
      long gidx = ((long)b * T + t) * H + hidx;
      float zx = gz[gidx] + rpre[0 * B * H + e];
      float ix = gi[gidx] + rpre[1 * B * H + e];
      float ox = go[gidx] + rpre[2 * B * H + e];
      float fx = gf[gidx] + rpre[3 * B * H + e];
      float ipre = CAP * tanhf(ix * (1.0f / CAP));
      float fpre = CAP * tanhf(fx * (1.0f / CAP));
      float zv = tanhf(zx);
      float ov = 1.0f / (1.0f + expf(-ox));
      float m_old = mst[e];
      float m_t = fmaxf(fpre + m_old, ipre);
      float ig = expf(ipre - m_t);
      float fg = expf(fpre - m_t + m_old);
      float c_t = fg * cst[e] + ig * zv;
      float n_t = fg * nst[e] + ig;
      float h_t = ov * (c_t / fmaxf(n_t, 1.0f));
      cst[e] = c_t; nst[e] = n_t; mst[e] = m_t; hst[e] = h_t;
      int head = hidx >> 7, dh = hidx & 127;
      hb[head * (16 * DH) + b * DH + dh] = f2bf(h_t);
    }
    grid_sync(cnt, gen, &s_gen);

    // -------- Phase C: multihead layernorm -> bf16 outb (16 x 1024, rows 8..15 zero)
    if (gwave < B * NH) {
      int b = gwave >> 3, head = gwave & 7;
      const float* hr = hst + (long)b * H + head * DH;
      float4 v = ((const float4*)hr)[lane];
      float s  = v.x + v.y + v.z + v.w;
      float ss = v.x * v.x + v.y * v.y + v.z * v.z + v.w * v.w;
      #pragma unroll
      for (int off = 16; off > 0; off >>= 1) {
        s  += __shfl_xor(s, off, 32);
        ss += __shfl_xor(ss, off, 32);
      }
      float mu = s * (1.0f / DH);
      float var = ss * (1.0f / DH) - mu * mu;
      float rstd = rsqrtf(var + EPS);
      int c0 = head * DH + lane * 4;
      float vals[4] = { v.x, v.y, v.z, v.w };
      #pragma unroll
      for (int k = 0; k < 4; ++k)
        outb[b * H + c0 + k] = f2bf((vals[k] - mu) * rstd * hln_w[c0 + k] + hln_b[c0 + k]);
    }
    grid_sync(cnt, gen, &s_gen);

    // -------- Phase D: up-projection  u = outb @ up_w + up_b  (16 x 2816)
    // stage the shared 16x1024 A-panel into LDS once per WG
    for (int cidx = tid; cidx < (16 * H) / 16; cidx += NTHR)
      *(v16bf*)&sAct[cidx * 16] = *(const v16bf*)&outb[cidx * 16];
    __syncthreads();
    #pragma unroll
    for (int r = 0; r < 2; ++r) {
      int nt = gwave + 128 * r;
      if (nt < P2PAD / 16) {
        int n0 = nt * 16;
        v8f acc = vzero;
        for (int k0 = 0; k0 < H; k0 += 32) {
          v16bf a  = *(const v16bf*)&sAct[rl * H + k0 + kh];
          v16bf bb;
          for (int s = 0; s < 16; ++s) bb[s] = upb[(long)(k0 + kh + s) * P2PAD + n0 + rl];
          acc = wmma_bf16(a, bb, acc);
        }
        if (lane < 16) {
          int col = n0 + lane;
          float bias = (col < P2) ? up_b[col] : 0.0f;
          #pragma unroll
          for (int v = 0; v < B; ++v)
            u[(long)v * P2PAD + col] = acc[v] + bias;
        }
      }
    }
    grid_sync(cnt, gen, &s_gen);

    // -------- Phase E: gact = u1 * gelu(u2)  -> bf16 (16 x 1376, rows 8..15 zero)
    for (int e = gtid; e < B * PPAD; e += NWG_SCAN * NTHR) {
      int b = e / PPAD, j = e - b * PPAD;
      float val = 0.0f;
      if (j < P) {
        float u1 = u[(long)b * P2PAD + j];
        float u2 = u[(long)b * P2PAD + P + j];
        val = u1 * (0.5f * u2 * (1.0f + erff(u2 * 0.70710678118654752f)));
      }
      gact[b * PPAD + j] = f2bf(val);
    }
    grid_sync(cnt, gen, &s_gen);

    // -------- Phase F: down-projection + residual -> out[b, t, :]
    // stage the shared 16x1376 A-panel into LDS once per WG
    for (int cidx = tid; cidx < (16 * PPAD) / 16; cidx += NTHR)
      *(v16bf*)&sAct[cidx * 16] = *(const v16bf*)&gact[cidx * 16];
    __syncthreads();
    if (gwave < D / 16) {
      int n0 = gwave * 16;
      v8f acc = vzero;
      for (int k0 = 0; k0 < PPAD; k0 += 32) {
        v16bf a  = *(const v16bf*)&sAct[rl * PPAD + k0 + kh];
        v16bf bb;
        for (int s = 0; s < 16; ++s) bb[s] = dnb[(long)(k0 + kh + s) * D + n0 + rl];
        acc = wmma_bf16(a, bb, acc);
      }
      if (lane < 16) {
        int col = n0 + lane;
        float db = down_b[col];
        #pragma unroll
        for (int v = 0; v < B; ++v) {
          long oidx = ((long)v * T + t) * D + col;
          out[oidx] = acc[v] + db + x[oidx];
        }
      }
    }
    // no global sync needed before next step's Phase A (disjoint buffers; the
    // cross-iteration LDS reuse of sAct is ordered by this WG's own barriers)
  }
}

// ---------------- host-side launcher ----------------
extern "C" void kernel_launch(void* const* d_in, const int* in_sizes, int n_in,
                              void* d_out, int out_size, void* d_ws, size_t ws_size,
                              hipStream_t stream) {
  const float* x     = (const float*)d_in[0];
  const float* ln_w  = (const float*)d_in[1];
  const float* ln_b  = (const float*)d_in[2];
  const float* hln_w = (const float*)d_in[3];
  const float* hln_b = (const float*)d_in[4];
  const float* Wz = (const float*)d_in[5];  const float* bz = (const float*)d_in[6];
  const float* Wi = (const float*)d_in[7];  const float* bi = (const float*)d_in[8];
  const float* Wo = (const float*)d_in[9];  const float* bo = (const float*)d_in[10];
  const float* Wf = (const float*)d_in[11]; const float* bfg = (const float*)d_in[12];
  const float* Rz = (const float*)d_in[13]; const float* rbz = (const float*)d_in[14];
  const float* Ri = (const float*)d_in[15]; const float* rbi = (const float*)d_in[16];
  const float* Ro = (const float*)d_in[17]; const float* rbo = (const float*)d_in[18];
  const float* Rf = (const float*)d_in[19]; const float* rbf = (const float*)d_in[20];
  const float* up_w   = (const float*)d_in[21];
  const float* up_b   = (const float*)d_in[22];
  const float* down_w = (const float*)d_in[23];
  const float* down_b = (const float*)d_in[24];
  float* out = (float*)d_out;

  char* ws = (char*)d_ws;
  bf16_t*  xn    = (bf16_t*)(ws + OFF_XN);
  float*   gates = (float*)(ws + OFF_G);
  bf16_t*  Wb    = (bf16_t*)(ws + OFF_WB);
  bf16_t*  Rb    = (bf16_t*)(ws + OFF_RB);
  bf16_t*  upb   = (bf16_t*)(ws + OFF_UPB);
  bf16_t*  dnb   = (bf16_t*)(ws + OFF_DNB);
  float*   cst   = (float*)(ws + OFF_C);
  float*   nst   = (float*)(ws + OFF_N);
  float*   hst   = (float*)(ws + OFF_HF);
  float*   mst   = (float*)(ws + OFF_M);
  bf16_t*  hb    = (bf16_t*)(ws + OFF_HB);
  float*   rpre  = (float*)(ws + OFF_RPRE);
  bf16_t*  outb  = (bf16_t*)(ws + OFF_OUTB);
  float*   u     = (float*)(ws + OFF_U);
  bf16_t*  gact  = (bf16_t*)(ws + OFF_GACT);
  unsigned* sync = (unsigned*)(ws + OFF_SYNC);

  // 1) deterministic re-init of all persistent/padded state
  k_init<<<256, NTHR, 0, stream>>>(cst, nst, hst, mst, hb, outb, gact, sync);
  // 2) bf16 weight conversion (with padding)
  k_prep<<<4096, NTHR, 0, stream>>>(Wz, Wi, Wo, Wf, Rz, Ri, Ro, Rf, up_w, down_w,
                                    Wb, Rb, upb, dnb);
  // 3) LayerNorm -> bf16 activations
  k_ln<<<B * T, NTHR, 0, stream>>>(x, ln_w, ln_b, xn);
  // 4) four input-projection GEMMs via WMMA (TDM double-buffered tiles)
  dim3 gg(B * T / 128, H / 128, 4);
  k_gemm_gates<<<gg, NTHR, 0, stream>>>(xn, Wb, bz, bi, bo, bfg, gates);
  // 5) persistent recurrent scan (16 WGs = one cluster's worth)
  k_scan<<<NWG_SCAN, NTHR, 0, stream>>>(
      gates + 0L * B * T * H, gates + 1L * B * T * H,
      gates + 2L * B * T * H, gates + 3L * B * T * H,
      Rb, rbz, rbi, rbo, rbf, upb, dnb,
      hln_w, hln_b, up_b, down_b, x, out,
      cst, nst, hst, mst, hb, rpre, outb, u, gact, sync);

  (void)in_sizes; (void)n_in; (void)out_size; (void)ws_size;
}
